// Seq2KVAttnDecoder_90348932038994
// MI455X (gfx1250) — matrix operand order repaired and checked by reading, not compiled, not run on hardware
//
#include <hip/hip_runtime.h>
#include <math.h>

typedef __bf16 bf16;
typedef __attribute__((ext_vector_type(16))) __bf16 v16bf;
typedef __attribute__((ext_vector_type(8)))  float  v8f;

#define Bc    32
#define SEc   256
#define HEc   512
#define HDc   1024
#define EINc  768
#define EDc   256
#define Nc    8192
#define Tc    20
#define KEc   1024
#define VEc   1024
#define MDc   256
#define NEGc  (-10000000.0f)
#define MAXVc (1000000000.0f)
#define EPSc  (1e-8f)
#define SPARSEc (0.8f)

// ---------------------------------------------------------------- WMMA core
__device__ __forceinline__ v8f wmma_bf16(v16bf a, v16bf b, v8f c) {
  // v_wmma_f32_16x16x32_bf16 (codegen-confirmed signature)
  return __builtin_amdgcn_wmma_f32_16x16x32_bf16(
      /*neg_a=*/false, a, /*neg_b=*/false, b,
      /*c_mod=*/(short)0, c, /*reuse_a=*/false, /*reuse_b=*/false);
}

// A fragment (16x32 bf16): per lane, elems 0..15 = base[0..15];
// caller adds m*lda + kc + 16*half.
__device__ __forceinline__ v16bf load_a_frag(const bf16* base) {
  v16bf a;
#pragma unroll
  for (int i = 0; i < 16; ++i) { a[i] = base[i]; }
  return a;
}

__device__ __forceinline__ float sigf(float x) { return 1.0f / (1.0f + __expf(-x)); }

// ---------------------------------------------------------------- utilities
__global__ void cvt_bf16_kernel(const float* __restrict__ src, bf16* __restrict__ dst, int n) {
  int i = blockIdx.x * 256 + threadIdx.x;
  int stride = gridDim.x * 256;
  for (; i < n; i += stride) dst[i] = (bf16)src[i];
}

__global__ void addvec_kernel(const float* a, const float* b, float* o, int n) {
  int i = blockIdx.x * 256 + threadIdx.x;
  if (i < n) o[i] = a[i] + b[i];
}

__global__ void fill_f32_kernel(float* p, float v, int n) {
  int i = blockIdx.x * 256 + threadIdx.x;
  if (i < n) p[i] = v;
}

__global__ void fill_bf16_kernel(bf16* p, float v, int n) {
  int i = blockIdx.x * 256 + threadIdx.x;
  if (i < n) p[i] = (bf16)v;
}

// h0/c0 = transpose(hidden,(1,0,2)).reshape(B,HD)
__global__ void init_state_kernel(const float* __restrict__ hidden,
                                  const float* __restrict__ cell,
                                  float* __restrict__ c, bf16* __restrict__ h_b) {
  int idx = blockIdx.x * 256 + threadIdx.x;           // 32*1024
  int b = idx >> 10, j = idx & 1023;
  int d = j >> 9, e = j & 511;
  size_t src = (size_t)d * Bc * HEc + (size_t)b * HEc + e;
  c[idx] = cell[src];
  h_b[idx] = (bf16)hidden[src];
}

// row L2 norms (dim=256), one block per row
__global__ void rownorm_kernel(const float* __restrict__ mat, float* __restrict__ out, int dim) {
  int row = blockIdx.x;
  __shared__ float red[256];
  float s = 0.f;
  for (int e = threadIdx.x; e < dim; e += 256) {
    float v = mat[(size_t)row * dim + e];
    s += v * v;
  }
  red[threadIdx.x] = s; __syncthreads();
  for (int k = 128; k > 0; k >>= 1) {
    if (threadIdx.x < k) red[threadIdx.x] += red[threadIdx.x + k];
    __syncthreads();
  }
  if (threadIdx.x == 0) out[row] = sqrtf(red[0]);
}

// ---------------------------------------------------------------- generic WMMA GEMM
// C[M=32 x N] = A0[32 x K0] @ W0[N x K0]^T (+ A1 @ W1^T) + bias + addmat, relu opt.
// One N-tile per wave; each wave carries BOTH M-tiles (two accumulators) so
// every weight (B) fragment is streamed from L2 exactly once.
__global__ void wmma_gemm_kernel(
    const bf16* __restrict__ A0, int lda0, const bf16* __restrict__ W0, int K0,
    const bf16* __restrict__ A1, int lda1, const bf16* __restrict__ W1, int K1,
    const float* __restrict__ bias, const float* __restrict__ addmat, int addld,
    float* __restrict__ Cout, int ldc, bf16* __restrict__ Cbf, int ldcb,
    int N, int relu) {
  int wave = threadIdx.x >> 5;
  int lane = threadIdx.x & 31;
  int ntile = blockIdx.x * (blockDim.x >> 5) + wave;
  if (ntile * 16 >= N) return;
  int half = lane >> 4;
  int l15  = lane & 15;
  int n = ntile * 16 + l15;    // W row / C col

  v8f acc[2];
  acc[0] = (v8f)(0.0f);
  acc[1] = (v8f)(0.0f);
  {
    const bf16* a0row = A0 + (size_t)l15 * lda0 + 16 * half;         // rows 0..15
    const bf16* a1row = A0 + (size_t)(16 + l15) * lda0 + 16 * half;  // rows 16..31
    const bf16* wrow  = W0 + (size_t)n * K0 + 16 * half;
    for (int kc = 0; kc < K0; kc += 32) {
      v16bf b  = *(const v16bf*)(wrow + kc);
      v16bf a0 = load_a_frag(a0row + kc);
      v16bf a1 = load_a_frag(a1row + kc);
      acc[0] = wmma_bf16(a0, b, acc[0]);
      acc[1] = wmma_bf16(a1, b, acc[1]);
    }
  }
  if (A1) {
    const bf16* a0row = A1 + (size_t)l15 * lda1 + 16 * half;
    const bf16* a1row = A1 + (size_t)(16 + l15) * lda1 + 16 * half;
    const bf16* wrow  = W1 + (size_t)n * K1 + 16 * half;
    for (int kc = 0; kc < K1; kc += 32) {
      v16bf b  = *(const v16bf*)(wrow + kc);
      v16bf a0 = load_a_frag(a0row + kc);
      v16bf a1 = load_a_frag(a1row + kc);
      acc[0] = wmma_bf16(a0, b, acc[0]);
      acc[1] = wmma_bf16(a1, b, acc[1]);
    }
  }
  // C layout: VGPR r -> row m = r + 8*half; col = lane&15 (ISA 7.12.2)
#pragma unroll
  for (int mt = 0; mt < 2; ++mt) {
#pragma unroll
    for (int r = 0; r < 8; ++r) {
      int row = mt * 16 + r + 8 * half;
      int col = ntile * 16 + l15;
      float v = acc[mt][r];
      if (bias)   v += bias[col];
      if (addmat) v += addmat[(size_t)row * addld + col];
      if (relu)   v = v > 0.f ? v : 0.f;
      if (Cout) Cout[(size_t)row * ldc + col] = v;
      if (Cbf)  Cbf[(size_t)row * ldcb + col] = (bf16)v;
    }
  }
}

// ---------------------------------------------------------------- LSTM pointwise
__global__ void lstm_pointwise_kernel(const float* __restrict__ g,
                                      float* __restrict__ c, bf16* __restrict__ h_b) {
  int idx = blockIdx.x * 256 + threadIdx.x;   // 32*1024
  int b = idx >> 10, j = idx & 1023;
  const float* gr = g + (size_t)b * 4 * HDc;
  float ii = sigf(gr[j]);
  float ff = sigf(gr[j + HDc]);
  float gg = tanhf(gr[j + 2 * HDc]);
  float oo = sigf(gr[j + 3 * HDc]);
  float c1 = ff * c[idx] + ii * gg;
  float h1 = oo * tanhf(c1);
  c[idx] = c1;
  h_b[idx] = (bf16)h1;
}

// ---------------------------------------------------------------- argmax + mask/emb update
__global__ void argmax_update_kernel(const float* __restrict__ out0, int t,
                                     float* __restrict__ nextids,
                                     float* __restrict__ lmask,
                                     const float* __restrict__ emb,
                                     bf16* __restrict__ x_b) {
  int b = blockIdx.x;
  const float* row = out0 + (size_t)b * Tc * Nc + (size_t)t * Nc;
  __shared__ float sv[256];
  __shared__ int   si[256];
  float best = -3.4e38f; int bi = Nc;
  for (int n = threadIdx.x; n < Nc; n += 256) {
    float v = row[n];
    if (v > best) { best = v; bi = n; }
  }
  sv[threadIdx.x] = best; si[threadIdx.x] = bi; __syncthreads();
  for (int s = 128; s > 0; s >>= 1) {
    if (threadIdx.x < s) {
      float v2 = sv[threadIdx.x + s]; int i2 = si[threadIdx.x + s];
      if (v2 > sv[threadIdx.x] || (v2 == sv[threadIdx.x] && i2 < si[threadIdx.x])) {
        sv[threadIdx.x] = v2; si[threadIdx.x] = i2;
      }
    }
    __syncthreads();
  }
  int idx = si[0];
  if (threadIdx.x == 0) {
    nextids[(size_t)b * Tc + t] = (float)idx;
    lmask[(size_t)b * Nc + idx] = NEGc;
    lmask[(size_t)b * Nc + 0]   = 0.0f;   // EOS
  }
  // lab = emb_table[idx] -> x[:, :ED]
  for (int e = threadIdx.x; e < EDc; e += 256)
    x_b[(size_t)b * EINc + e] = (bf16)emb[(size_t)idx * EDc + e];
}

// ---------------------------------------------------------------- attention (score+softmax+attn)
__global__ void attn_kernel(const float* __restrict__ enc,
                            const unsigned char* __restrict__ emask,
                            const float* __restrict__ q,
                            bf16* __restrict__ x_b) {
  int b = blockIdx.x, tid = threadIdx.x;
  __shared__ float sq[HEc];
  __shared__ float sw[SEc];
  __shared__ float red[256];
  for (int e = tid; e < HEc; e += 256) sq[e] = q[(size_t)b * HEc + e];
  __syncthreads();
  const float* eb = enc + (size_t)b * SEc * HEc;
  int s = tid;                                  // SEc == 256 == blockDim
  float sc = 0.f;
  for (int e = 0; e < HEc; ++e) sc += eb[(size_t)s * HEc + e] * sq[e];
  sc += emask[(size_t)b * SEc + s] ? 0.0f : -MAXVc;
  red[tid] = sc; __syncthreads();
  for (int k = 128; k > 0; k >>= 1) {
    if (tid < k) red[tid] = fmaxf(red[tid], red[tid + k]);
    __syncthreads();
  }
  float mx = red[0]; __syncthreads();
  float ex = __expf(sc - mx);
  red[tid] = ex; __syncthreads();
  for (int k = 128; k > 0; k >>= 1) {
    if (tid < k) red[tid] += red[tid + k];
    __syncthreads();
  }
  float inv = 1.0f / red[0];
  sw[tid] = ex * inv; __syncthreads();
  for (int e = tid; e < HEc; e += 256) {
    float a = 0.f;
    for (int s2 = 0; s2 < SEc; ++s2) a += sw[s2] * eb[(size_t)s2 * HEc + e];
    x_b[(size_t)b * EINc + EDc + e] = (bf16)(eb[e] + a);   // sent + attn
  }
}

// ---------------------------------------------------------------- ksim (running max of cos)
__global__ void ksim_kernel(const float* __restrict__ mq,
                            const float* __restrict__ memk,
                            const float* __restrict__ knorm,
                            float* __restrict__ keysim) {
  int b = blockIdx.x, tid = threadIdx.x;
  __shared__ float smq[MDc];
  __shared__ float red[256];
  float v = mq[(size_t)b * MDc + tid];
  smq[tid] = v; red[tid] = v * v; __syncthreads();
  for (int k = 128; k > 0; k >>= 1) {
    if (tid < k) red[tid] += red[tid + k];
    __syncthreads();
  }
  float nq = sqrtf(red[0]);
  for (int k = tid; k < KEc; k += 256) {
    const float* kr = memk + ((size_t)b * KEc + k) * MDc;
    float d = 0.f;
    for (int e = 0; e < MDc; ++e) d += kr[e] * smq[e];
    float cs = d / fmaxf(nq * knorm[(size_t)b * KEc + k], EPSc);
    size_t o = (size_t)b * KEc + k;
    keysim[o] = fmaxf(keysim[o], cs);
  }
}

// ---------------------------------------------------------------- mw pass A: rowmax over v (WMMA)
// One wave handles TWO k-tiles (A frags resident) so each mem_v (B) fragment
// streamed from L2 feeds two WMMA chains.
__global__ void rowmax_kernel(const bf16* __restrict__ memk_b,
                              const bf16* __restrict__ memv_b,
                              const float* __restrict__ knorm,
                              const float* __restrict__ vnorm,
                              float* __restrict__ rowmax) {
  int wave = threadIdx.x >> 5;
  int lane = threadIdx.x & 31;
  int tile = blockIdx.x * (blockDim.x >> 5) + wave;   // 32*32 tiles
  int bb = tile >> 5, sub = tile & 31;
  int half = lane >> 4, l15 = lane & 15;

  v16bf afr[2][8];
#pragma unroll
  for (int p = 0; p < 2; ++p) {
    int ktile = sub * 2 + p;
    const bf16* arow = memk_b + ((size_t)bb * KEc + ktile * 16 + l15) * MDc + 16 * half;
#pragma unroll
    for (int kc = 0; kc < 8; ++kc) afr[p][kc] = load_a_frag(arow + kc * 32);
  }

  float rmax[2][8];
#pragma unroll
  for (int p = 0; p < 2; ++p)
#pragma unroll
    for (int r = 0; r < 8; ++r) rmax[p][r] = -3.4e38f;

  for (int vt = 0; vt < 64; ++vt) {
    const bf16* vrow = memv_b + ((size_t)bb * VEc + vt * 16 + l15) * MDc + 16 * half;
    v8f acc0 = (v8f)(0.0f);
    v8f acc1 = (v8f)(0.0f);
#pragma unroll
    for (int kc = 0; kc < 8; ++kc) {
      v16bf bfr = *(const v16bf*)(vrow + kc * 32);
      acc0 = wmma_bf16(afr[0][kc], bfr, acc0);
      acc1 = wmma_bf16(afr[1][kc], bfr, acc1);
    }
    int vn = vt * 16 + l15;
    float nv = vnorm[(size_t)bb * VEc + vn];
#pragma unroll
    for (int r = 0; r < 8; ++r) {
      int km0 = (sub * 2 + 0) * 16 + r + 8 * half;
      int km1 = (sub * 2 + 1) * 16 + r + 8 * half;
      float c0 = acc0[r] / fmaxf(knorm[(size_t)bb * KEc + km0] * nv, EPSc);
      float c1 = acc1[r] / fmaxf(knorm[(size_t)bb * KEc + km1] * nv, EPSc);
      rmax[0][r] = fmaxf(rmax[0][r], c0);
      rmax[1][r] = fmaxf(rmax[1][r], c1);
    }
  }
  // reduce across the 16 lanes of each half (same k row, different v)
#pragma unroll
  for (int off = 1; off < 16; off <<= 1) {
#pragma unroll
    for (int p = 0; p < 2; ++p)
#pragma unroll
      for (int r = 0; r < 8; ++r)
        rmax[p][r] = fmaxf(rmax[p][r], __shfl_xor(rmax[p][r], off, 32));
  }
  if (l15 == 0) {
#pragma unroll
    for (int p = 0; p < 2; ++p)
#pragma unroll
      for (int r = 0; r < 8; ++r)
        rowmax[(size_t)bb * KEc + (sub * 2 + p) * 16 + r + 8 * half] = rmax[p][r];
  }
}

// ---------------------------------------------------------------- mw pass B: sparsify + val_sim (WMMA)
// One wave handles TWO v-tiles (B frags resident) so each mem_k (A) fragment
// streamed from L2 feeds two WMMA chains. Accumulation order matches pass A
// bitwise so the ">= rowmax" sparsity test is exact.
__global__ void valsim_kernel(const bf16* __restrict__ memk_b,
                              const bf16* __restrict__ memv_b,
                              const float* __restrict__ knorm,
                              const float* __restrict__ vnorm,
                              const float* __restrict__ rowmax,
                              const float* __restrict__ keysim,
                              const unsigned char* __restrict__ vmask,
                              float* __restrict__ valout) {
  int wave = threadIdx.x >> 5;
  int lane = threadIdx.x & 31;
  int tile = blockIdx.x * (blockDim.x >> 5) + wave;   // 32*32 tiles
  int bb = tile >> 5, sub = tile & 31;
  int half = lane >> 4, l15 = lane & 15;

  v16bf bfr[2][8];
  float nv[2];
#pragma unroll
  for (int p = 0; p < 2; ++p) {
    int vn = (sub * 2 + p) * 16 + l15;
    const bf16* vrow = memv_b + ((size_t)bb * VEc + vn) * MDc + 16 * half;
#pragma unroll
    for (int kc = 0; kc < 8; ++kc) bfr[p][kc] = *(const v16bf*)(vrow + kc * 32);
    nv[p] = vnorm[(size_t)bb * VEc + vn];
  }

  float accv[2] = {0.f, 0.f};
  for (int kt = 0; kt < 64; ++kt) {
    const bf16* arow = memk_b + ((size_t)bb * KEc + kt * 16 + l15) * MDc + 16 * half;
    v8f acc0 = (v8f)(0.0f);
    v8f acc1 = (v8f)(0.0f);
#pragma unroll
    for (int kc = 0; kc < 8; ++kc) {
      v16bf a = load_a_frag(arow + kc * 32);
      acc0 = wmma_bf16(a, bfr[0][kc], acc0);
      acc1 = wmma_bf16(a, bfr[1][kc], acc1);
    }
#pragma unroll
    for (int r = 0; r < 8; ++r) {
      int km = kt * 16 + r + 8 * half;
      float rm = rowmax[(size_t)bb * KEc + km];
      float nk = knorm[(size_t)bb * KEc + km];
      float ks = keysim[(size_t)bb * KEc + km];
      ks = ks < 0.f ? 0.f : ks;
      float c0 = acc0[r] / fmaxf(nk * nv[0], EPSc);
      float c1 = acc1[r] / fmaxf(nk * nv[1], EPSc);
      if (c0 >= rm && c0 >= SPARSEc) accv[0] += ks * c0;
      if (c1 >= rm && c1 >= SPARSEc) accv[1] += ks * c1;
    }
  }
#pragma unroll
  for (int p = 0; p < 2; ++p) {
    float a = accv[p] + __shfl_xor(accv[p], 16, 32);  // combine the two row-halves
    if (lane < 16) {
      size_t o = (size_t)bb * VEc + (sub * 2 + p) * 16 + l15;
      valout[o] = vmask[o] ? a : -1.0f;
    }
  }
}

// ---------------------------------------------------------------- key_out
__global__ void keyout_kernel(const float* __restrict__ keysim,
                              const unsigned char* __restrict__ kmask,
                              float* __restrict__ keyout) {
  int idx = blockIdx.x * 256 + threadIdx.x;   // 32*1024
  float ks = keysim[idx];
  ks = ks < 0.f ? 0.f : ks;
  keyout[idx] = kmask[idx] ? ks : -1.0f;
}

// ================================================================ host
extern "C" void kernel_launch(void* const* d_in, const int* in_sizes, int n_in,
                              void* d_out_, int out_size, void* d_ws, size_t ws_size,
                              hipStream_t stream) {
  const float* enc    = (const float*)d_in[0];
  const float* hidden = (const float*)d_in[1];
  const float* cell   = (const float*)d_in[2];
  const unsigned char* emask = (const unsigned char*)d_in[4];
  const unsigned char* kmask = (const unsigned char*)d_in[5];
  const float* memk   = (const float*)d_in[6];
  const unsigned char* vmask = (const unsigned char*)d_in[7];
  const float* memv   = (const float*)d_in[8];
  const float* W_ih   = (const float*)d_in[9];
  const float* W_hh   = (const float*)d_in[10];
  const float* b_ih   = (const float*)d_in[11];
  const float* b_hh   = (const float*)d_in[12];
  const float* W1     = (const float*)d_in[13];
  const float* b1     = (const float*)d_in[14];
  const float* W2     = (const float*)d_in[15];
  const float* b2     = (const float*)d_in[16];
  const float* emb    = (const float*)d_in[17];
  const float* dW     = (const float*)d_in[18];
  const float* db     = (const float*)d_in[19];
  const float* rW     = (const float*)d_in[20];
  const float* rb     = (const float*)d_in[21];
  float* out = (float*)d_out_;

  const size_t OUT0 = 0;                               // outputs (B,T,N)
  const size_t OUT1 = (size_t)Bc * Tc * Nc;            // next_ids (B,T)
  const size_t OUT2 = OUT1 + (size_t)Bc * Tc;          // key_out (B,KE)
  const size_t OUT3 = OUT2 + (size_t)Bc * KEc;         // val_out (B,VE)

  // workspace bump allocator (256B aligned)
  char* wsp = (char*)d_ws;
  size_t off = 0;
  auto take = [&](size_t bytes) -> void* {
    void* p = wsp + off;
    off += (bytes + 255) & ~(size_t)255;
    return p;
  };
  bf16* Wih_b  = (bf16*)take((size_t)4 * HDc * EINc * 2);
  bf16* Whh_b  = (bf16*)take((size_t)4 * HDc * HDc  * 2);
  bf16* W1_b   = (bf16*)take((size_t)HDc * HDc * 2);
  bf16* W2_b   = (bf16*)take((size_t)Nc * HDc * 2);
  bf16* dW_b   = (bf16*)take((size_t)HEc * EDc * 2);
  bf16* rW_b   = (bf16*)take((size_t)MDc * EINc * 2);
  bf16* memk_b = (bf16*)take((size_t)Bc * KEc * MDc * 2);
  bf16* memv_b = (bf16*)take((size_t)Bc * VEc * MDc * 2);
  float* bias_g = (float*)take((size_t)4 * HDc * 4);
  bf16* h_b    = (bf16*)take((size_t)Bc * HDc * 2);
  bf16* x_b    = (bf16*)take((size_t)Bc * EINc * 2);
  bf16* hid_b  = (bf16*)take((size_t)Bc * HDc * 2);
  float* cst   = (float*)take((size_t)Bc * HDc * 4);
  float* g     = (float*)take((size_t)Bc * 4 * HDc * 4);
  float* q     = (float*)take((size_t)Bc * HEc * 4);
  float* mq    = (float*)take((size_t)Bc * MDc * 4);
  float* lmask = (float*)take((size_t)Bc * Nc * 4);
  float* keysim= (float*)take((size_t)Bc * KEc * 4);
  float* knorm = (float*)take((size_t)Bc * KEc * 4);
  float* vnorm = (float*)take((size_t)Bc * VEc * 4);
  float* rowmx = (float*)take((size_t)Bc * KEc * 4);
  (void)ws_size; (void)in_sizes; (void)n_in; (void)out_size;

  auto cvt = [&](const float* s, bf16* d, size_t n) {
    int blocks = (int)((n + 255) / 256);
    if (blocks > 4096) blocks = 4096;
    cvt_bf16_kernel<<<blocks, 256, 0, stream>>>(s, d, (int)n);
  };
  cvt(W_ih, Wih_b, (size_t)4 * HDc * EINc);
  cvt(W_hh, Whh_b, (size_t)4 * HDc * HDc);
  cvt(W1,   W1_b,  (size_t)HDc * HDc);
  cvt(W2,   W2_b,  (size_t)Nc * HDc);
  cvt(dW,   dW_b,  (size_t)HEc * EDc);
  cvt(rW,   rW_b,  (size_t)MDc * EINc);
  cvt(memk, memk_b,(size_t)Bc * KEc * MDc);
  cvt(memv, memv_b,(size_t)Bc * VEc * MDc);

  addvec_kernel<<<(4 * HDc) / 256, 256, 0, stream>>>(b_ih, b_hh, bias_g, 4 * HDc);
  init_state_kernel<<<(Bc * HDc) / 256, 256, 0, stream>>>(hidden, cell, cst, h_b);
  fill_bf16_kernel<<<(Bc * EINc) / 256, 256, 0, stream>>>(x_b, 0.0f, Bc * EINc);
  fill_f32_kernel<<<(Bc * Nc) / 256, 256, 0, stream>>>(lmask, 0.0f, Bc * Nc);
  fill_f32_kernel<<<(Bc * KEc) / 256, 256, 0, stream>>>(keysim, -3.0e38f, Bc * KEc);
  rownorm_kernel<<<Bc * KEc, 256, 0, stream>>>(memk, knorm, MDc);
  rownorm_kernel<<<Bc * VEc, 256, 0, stream>>>(memv, vnorm, MDc);

  auto gemm = [&](const bf16* A0, int lda0, const bf16* W0p, int K0,
                  const bf16* A1, int lda1, const bf16* W1p, int K1,
                  const float* bias, const float* addm, int addld,
                  float* C, int ldc, bf16* Cb, int ldcb, int N, int relu) {
    int tiles = N / 16;                       // one N-tile per wave, both M-tiles
    int blocks = (tiles + 7) / 8;
    wmma_gemm_kernel<<<blocks, 256, 0, stream>>>(
        A0, lda0, W0p, K0, A1, lda1, W1p, K1,
        bias, addm, addld, C, ldc, Cb, ldcb, N, relu);
  };

  for (int t = 0; t < Tc; ++t) {
    // gates = x@Wih^T + h@Whh^T + (b_ih+b_hh)
    gemm(x_b, EINc, Wih_b, EINc, h_b, HDc, Whh_b, HDc,
         bias_g, nullptr, 0, g, 4 * HDc, nullptr, 0, 4 * HDc, 0);
    lstm_pointwise_kernel<<<(Bc * HDc) / 256, 256, 0, stream>>>(g, cst, h_b);
    // hid = relu(h@W1^T + b1)
    gemm(h_b, HDc, W1_b, HDc, nullptr, 0, nullptr, 0,
         b1, nullptr, 0, nullptr, 0, hid_b, HDc, HDc, 1);
    // logits = hid@W2^T + b2 + lmask  -> directly into d_out (B,T,N)
    gemm(hid_b, HDc, W2_b, HDc, nullptr, 0, nullptr, 0,
         b2, lmask, Nc, out + OUT0 + (size_t)t * Nc, Tc * Nc, nullptr, 0, Nc, 0);
    argmax_update_kernel<<<Bc, 256, 0, stream>>>(out + OUT0, t, out + OUT1, lmask, emb, x_b);
    // q = lab@dense_W^T + db  (lab = first ED cols of x)
    gemm(x_b, EINc, dW_b, EDc, nullptr, 0, nullptr, 0,
         db, nullptr, 0, q, HEc, nullptr, 0, HEc, 0);
    attn_kernel<<<Bc, 256, 0, stream>>>(enc, emask, q, x_b);
    // mq = x@read_W^T + rb
    gemm(x_b, EINc, rW_b, EINc, nullptr, 0, nullptr, 0,
         rb, nullptr, 0, mq, MDc, nullptr, 0, MDc, 0);
    ksim_kernel<<<Bc, 256, 0, stream>>>(mq, memk, knorm, keysim);
  }

  // mw = cos(mem_k, mem_v): rowmax pass, then sparsify+accumulate pass (bitwise-identical math)
  rowmax_kernel<<<(Bc * 32) / 8, 256, 0, stream>>>(memk_b, memv_b, knorm, vnorm, rowmx);
  valsim_kernel<<<(Bc * 32) / 8, 256, 0, stream>>>(memk_b, memv_b, knorm, vnorm,
                                                   rowmx, keysim, vmask, out + OUT3);
  keyout_kernel<<<(Bc * KEc) / 256, 256, 0, stream>>>(keysim, kmask, out + OUT2);
}
